// Net_29437705847344
// MI455X (gfx1250) — matrix operand order repaired and checked by reading, not compiled, not run on hardware
//
#include <hip/hip_runtime.h>
#include <hip/hip_bf16.h>
#include <math.h>

#define N_NODES 50000
#define N_EDGES 800000
#define IN_DIM  9
#define HID     128
#define NGROUP  256
#define OUT_DIM 10
#define EPSV    1e-5f
#define RAT_ROWS 256

typedef __attribute__((ext_vector_type(16))) __bf16 v16bf;
typedef __attribute__((ext_vector_type(8)))  float  v8f;
typedef __attribute__((ext_vector_type(4)))  unsigned int u32x4;
typedef __attribute__((ext_vector_type(8)))  int i32x8;
typedef __attribute__((ext_vector_type(4)))  int i32x4;

union Frag { v16bf v; uint4 u[2]; };

// ---------------------------------------------------------------- utilities
__global__ void k_fill(float* p, float val, int n) {
  int i = blockIdx.x * 256 + threadIdx.x;
  if (i < n) p[i] = val;
}

__global__ void k_deg(const int* __restrict__ dst, float* __restrict__ deg, int E) {
  int e = blockIdx.x * 256 + threadIdx.x;
  if (e < E) atomicAdd(&deg[dst[e]], 1.0f);
}

__global__ void k_dinv(const float* __restrict__ deg, float* __restrict__ dinv, int n) {
  int i = blockIdx.x * 256 + threadIdx.x;
  if (i < n) dinv[i] = rsqrtf(deg[i]);
}

// ------------------------------------------------- layer-0 GEMM (K=9, tiny)
__global__ void k_gemm_in(const float* __restrict__ x, const float* __restrict__ W,
                          float* __restrict__ hh, int n) {
  int idx = blockIdx.x * 256 + threadIdx.x;
  if (idx >= n * HID) return;
  int i = idx / HID, c = idx % HID;
  const float* xr = x + (size_t)i * IN_DIM;
  float s = 0.0f;
#pragma unroll
  for (int k = 0; k < IN_DIM; ++k) s = fmaf(xr[k], W[k * HID + c], s);
  hh[idx] = s;
}

// --------------------------------------------- W[k][n] -> Wt[n][k] in bf16
__global__ void k_wt(const float* __restrict__ W, __hip_bfloat16* __restrict__ Wt) {
  int idx = blockIdx.x * 256 + threadIdx.x;
  if (idx >= HID * HID) return;
  int nn = idx / HID, kk = idx % HID;
  Wt[idx] = __float2bfloat16(W[kk * HID + nn]);
}

// --------------------------------------------------- WMMA GEMM: A[M,128]@W
// block = 256 threads = 8 waves, each wave computes a 16x128 row strip.
// Wt (W^T, bf16, row-major [n][k]) staged in LDS via the Tensor Data Mover;
// B frags come from LDS via ds_load_b128, A frags via global_load_b128.
__global__ __launch_bounds__(256) void k_gemm_wmma(
    const __hip_bfloat16* __restrict__ A,   // [M][128] bf16, row major
    const __hip_bfloat16* __restrict__ Wt,  // [128][128] = W^T, bf16
    float* __restrict__ out, int M) {
  __shared__ __hip_bfloat16 sW[HID * HID];  // 32 KB

  int tid = threadIdx.x;
  int wave = tid >> 5;
  int lane = tid & 31;
  int lh   = lane & 15;
  int khalf = lane >> 4;   // 0 or 1

  int rowbase = blockIdx.x * 128 + wave * 16;
  int arow = rowbase + lh;
  int arowc = arow < M ? arow : (M - 1);    // clamp: keep EXEC all-ones for WMMA
  const uint4* Arow = (const uint4*)(A + (size_t)arowc * HID);
  __builtin_prefetch(Arow, 0, 3);           // global_prefetch_b8, overlaps W staging

#if __has_builtin(__builtin_amdgcn_tensor_load_to_lds) && \
    __has_builtin(__builtin_amdgcn_s_wait_tensorcnt)
  // ---- Tensor Data Mover: DMA the whole 128x128 bf16 Wt tile into LDS ----
  if (tid < 32) {  // one wave issues the descriptor
    unsigned int lds_off = (unsigned int)(uintptr_t)(&sW[0]);   // flat->LDS: low 32 bits
    unsigned long long ga = (unsigned long long)(uintptr_t)Wt;
    u32x4 g0;
    g0[0] = 1u;                                           // count=1, user D#
    g0[1] = lds_off;                                      // lds_addr
    g0[2] = (unsigned int)(ga & 0xFFFFFFFFull);           // global_addr[31:0]
    g0[3] = (unsigned int)((ga >> 32) & 0x1FFFFFFull)     // global_addr[56:32]
            | (2u << 30);                                 // type = 2 ("image")
    i32x8 g1;
    g1[0] = (1 << 16);            // workgroup_mask=0, data_size=1 (2 bytes)
    g1[1] = (HID << 16);          // tensor_dim0[15:0] in bits[63:48]
    g1[2] = (HID << 16);          // tensor_dim0[31:16]=0, tensor_dim1[15:0]=128
    g1[3] = (HID << 16);          // tensor_dim1[31:16]=0, tile_dim0=128
    g1[4] = HID;                  // tile_dim1=128, tile_dim2=0
    g1[5] = HID;                  // tensor_dim0_stride[31:0] = 128 elements
    g1[6] = 0;
    g1[7] = 0;
    i32x4 gz; gz[0] = 0; gz[1] = 0; gz[2] = 0; gz[3] = 0;  // 2-D tensor: groups 2/3 unused
    i32x8 gz8;
    gz8[0] = 0; gz8[1] = 0; gz8[2] = 0; gz8[3] = 0;
    gz8[4] = 0; gz8[5] = 0; gz8[6] = 0; gz8[7] = 0;
    __builtin_amdgcn_tensor_load_to_lds(g0, g1, gz, gz, gz8, 0);  // 6-arg flavor
    __builtin_amdgcn_s_wait_tensorcnt(0);
  }
#else
  // ---- fallback: cooperative vector copy ----
  {
    const uint4* gsrc = (const uint4*)Wt;
    uint4* ldst = (uint4*)sW;
    for (int i = tid; i < HID * HID / 8; i += 256) ldst[i] = gsrc[i];
  }
#endif
  __syncthreads();

  v8f acc[8];
#pragma unroll
  for (int t = 0; t < 8; ++t)
#pragma unroll
    for (int r = 0; r < 8; ++r) acc[t][r] = 0.0f;

  for (int k0 = 0; k0 < HID; k0 += 32) {
    // A fragment (16x32 bf16): lane<16 -> K {k0..k0+7, k0+16..k0+23}; lane>=16 shifts by 8
    Frag a;
    a.u[0] = Arow[(k0 + khalf * 8) >> 3];
    a.u[1] = Arow[(k0 + 16 + khalf * 8) >> 3];

    // Batch-load all 8 B fragments for this K-step into distinct registers so
    // the ds_load_b128s issue as one clause and the 8 WMMAs run back-to-back.
    Frag b[8];
    int kb = k0 + khalf * 8;      // lanes<16: K {k0..k0+7, k0+16..k0+23}; lanes>=16 shift by 8
#pragma unroll
    for (int t = 0; t < 8; ++t) {
      const uint4* Bcol = (const uint4*)(sW + (t * 16 + lh) * HID);
      b[t].u[0] = Bcol[kb >> 3];
      b[t].u[1] = Bcol[(kb + 16) >> 3];
    }
#pragma unroll
    for (int t = 0; t < 8; ++t) {
      acc[t] = __builtin_amdgcn_wmma_f32_16x16x32_bf16(
          false, a.v, false, b[t].v, (short)0, acc[t], false, false);
    }
  }

  if (rowbase < M) {
#pragma unroll
    for (int t = 0; t < 8; ++t) {
      int col = t * 16 + lh;
#pragma unroll
      for (int r = 0; r < 8; ++r) {
        int row = rowbase + r + khalf * 8;
        if (row < M) out[(size_t)row * HID + col] = acc[t][r];
      }
    }
  }
}

// --------------------------------------------------------- GCN aggregation
__global__ void k_init_agg(const float* __restrict__ hh, const float* __restrict__ dinv,
                           const float* __restrict__ b, float* __restrict__ agg, int n) {
  int idx = blockIdx.x * 256 + threadIdx.x;
  if (idx >= n * HID) return;
  int i = idx / HID, c = idx % HID;
  float di = dinv[i];
  agg[idx] = hh[idx] * di * di + b[c];
}

__global__ void k_edge_agg(const int* __restrict__ src, const int* __restrict__ dst,
                           const float* __restrict__ dinv, const float* __restrict__ hh,
                           float* __restrict__ agg, int E) {
  long long gid = (long long)blockIdx.x * 256 + threadIdx.x;
  int e = (int)(gid >> 5);
  if (e >= E) return;
  int p = (int)(gid & 31);
  int s = src[e], d = dst[e];
  float coef = dinv[s] * dinv[d];
  const float4 hv = ((const float4*)(hh + (size_t)s * HID))[p];
  float* ap = agg + (size_t)d * HID + p * 4;
  atomicAdd(ap + 0, hv.x * coef);
  atomicAdd(ap + 1, hv.y * coef);
  atomicAdd(ap + 2, hv.z * coef);
  atomicAdd(ap + 3, hv.w * coef);
}

// ------------------------------------- rational activation + BN stats pass
__global__ __launch_bounds__(128) void k_rational(
    const float* __restrict__ agg, const float* __restrict__ num,
    const float* __restrict__ den, const int* __restrict__ assign,
    float* __restrict__ act, float* __restrict__ stats, int n) {
  int c = threadIdx.x;
  int r0 = blockIdx.x * RAT_ROWS;
  int cl = assign[c];
  float a0 = num[cl * 6 + 0], a1 = num[cl * 6 + 1], a2 = num[cl * 6 + 2];
  float a3 = num[cl * 6 + 3], a4 = num[cl * 6 + 4], a5 = num[cl * 6 + 5];
  float q0 = den[cl * 4 + 0], q1 = den[cl * 4 + 1], q2 = den[cl * 4 + 2], q3 = den[cl * 4 + 3];
  float s = 0.0f, s2 = 0.0f;
  int rend = r0 + RAT_ROWS; if (rend > n) rend = n;
  for (int i = r0; i < rend; ++i) {
    float x = agg[(size_t)i * HID + c];
    float P = fmaf(fmaf(fmaf(fmaf(fmaf(a5, x, a4), x, a3), x, a2), x, a1), x, a0);
    float Q = fmaf(fmaf(fmaf(q3, x, q2), x, q1), x, q0) * x;
    float r = P / (1.0f + fabsf(Q));
    act[(size_t)i * HID + c] = r;
    s += r; s2 += r * r;
  }
  atomicAdd(&stats[c], s);
  atomicAdd(&stats[HID + c], s2);
}

// ----------------------------------------- BN apply + bf16 copy for WMMA
__global__ void k_bn(float* __restrict__ act, const float* __restrict__ stats,
                     const float* __restrict__ g, const float* __restrict__ be,
                     __hip_bfloat16* __restrict__ abf, int n) {
  int idx = blockIdx.x * 256 + threadIdx.x;
  if (idx >= n * HID) return;
  int c = idx % HID;
  float inv_n = 1.0f / (float)n;
  float m = stats[c] * inv_n;
  float v = stats[HID + c] * inv_n - m * m;
  float y = g[c] * (act[idx] - m) * rsqrtf(v + EPSV) + be[c];
  act[idx] = y;
  abf[idx] = __float2bfloat16(y);
}

// ------------------------------------------------------------- pooling/MLP
__global__ void k_counts(const int* __restrict__ batch, float* __restrict__ counts, int n) {
  int i = blockIdx.x * 256 + threadIdx.x;
  if (i < n) atomicAdd(&counts[batch[i]], 1.0f);
}

__global__ void k_pool(const int* __restrict__ batch, const float* __restrict__ act,
                       float* __restrict__ pooled, int n) {
  int idx = blockIdx.x * 256 + threadIdx.x;
  if (idx >= n * HID) return;
  int i = idx / HID, c = idx % HID;
  atomicAdd(&pooled[(size_t)batch[i] * HID + c], act[idx]);
}

__global__ void k_pdiv(float* __restrict__ pooled, const float* __restrict__ counts) {
  int idx = blockIdx.x * 256 + threadIdx.x;
  if (idx >= NGROUP * HID) return;
  pooled[idx] /= fmaxf(counts[idx / HID], 1.0f);
}

__global__ void k_mlp_gelu(const float* __restrict__ in, const float* __restrict__ W,
                           const float* __restrict__ b, float* __restrict__ out) {
  int idx = blockIdx.x * 256 + threadIdx.x;
  if (idx >= NGROUP * HID) return;
  int gI = idx / HID, c = idx % HID;
  const float* r = in + (size_t)gI * HID;
  float s = b[c];
  for (int k = 0; k < HID; ++k) s = fmaf(r[k], W[k * HID + c], s);
  out[idx] = 0.5f * s * (1.0f + erff(s * 0.70710678118f));
}

__global__ void k_mlp_out(const float* __restrict__ in, const float* __restrict__ W,
                          const float* __restrict__ b, float* __restrict__ out) {
  int idx = blockIdx.x * 256 + threadIdx.x;
  if (idx >= NGROUP * OUT_DIM) return;
  int gI = idx / OUT_DIM, o = idx % OUT_DIM;
  const float* r = in + (size_t)gI * HID;
  float s = b[o];
  for (int k = 0; k < HID; ++k) s = fmaf(r[k], W[k * OUT_DIM + o], s);
  out[idx] = s;
}

// ------------------------------------------------------------------ driver
static inline int cdiv_ll(long long a, long long b) { return (int)((a + b - 1) / b); }

extern "C" void kernel_launch(void* const* d_in, const int* in_sizes, int n_in,
                              void* d_out, int out_size, void* d_ws, size_t ws_size,
                              hipStream_t stream) {
  const float* x      = (const float*)d_in[0];
  const int*   ei     = (const int*)d_in[1];
  const int*   batch  = (const int*)d_in[2];
  const float* W[4]   = {(const float*)d_in[3],  (const float*)d_in[7],
                         (const float*)d_in[11], (const float*)d_in[15]};
  const float* bb[4]  = {(const float*)d_in[4],  (const float*)d_in[8],
                         (const float*)d_in[12], (const float*)d_in[16]};
  const float* gg[4]  = {(const float*)d_in[5],  (const float*)d_in[9],
                         (const float*)d_in[13], (const float*)d_in[17]};
  const float* beta[4]= {(const float*)d_in[6],  (const float*)d_in[10],
                         (const float*)d_in[14], (const float*)d_in[18]};
  const float* rnum   = (const float*)d_in[19];
  const float* rden   = (const float*)d_in[20];
  const int*   assign = (const int*)d_in[21];
  const float* HW1 = (const float*)d_in[22]; const float* Hb1 = (const float*)d_in[23];
  const float* HW2 = (const float*)d_in[24]; const float* Hb2 = (const float*)d_in[25];
  const float* HW3 = (const float*)d_in[26]; const float* Hb3 = (const float*)d_in[27];
  float* out = (float*)d_out;

  float* ws = (float*)d_ws;
  size_t off = 0;
  float* HH  = ws + off; off += (size_t)N_NODES * HID;
  float* AGG = ws + off; off += (size_t)N_NODES * HID;
  float* ACT = ws + off; off += (size_t)N_NODES * HID;
  __hip_bfloat16* Abf = (__hip_bfloat16*)(ws + off); off += (size_t)N_NODES * HID / 2;
  __hip_bfloat16* Wt  = (__hip_bfloat16*)(ws + off); off += HID * HID / 2;
  float* deg    = ws + off; off += N_NODES;
  float* dinv   = ws + off; off += N_NODES;
  float* stats  = ws + off; off += 2 * HID;
  float* pooled = ws + off; off += NGROUP * HID;
  float* counts = ws + off; off += NGROUP;
  float* z1     = ws + off; off += NGROUP * HID;
  float* z2     = ws + off; off += NGROUP * HID;

  const int* srcI = ei;
  const int* dstI = ei + N_EDGES;
  const long long NH = (long long)N_NODES * HID;

  k_fill<<<cdiv_ll(N_NODES, 256), 256, 0, stream>>>(deg, 1.0f, N_NODES);
  k_deg<<<cdiv_ll(N_EDGES, 256), 256, 0, stream>>>(dstI, deg, N_EDGES);
  k_dinv<<<cdiv_ll(N_NODES, 256), 256, 0, stream>>>(deg, dinv, N_NODES);

  k_gemm_in<<<cdiv_ll(NH, 256), 256, 0, stream>>>(x, W[0], HH, N_NODES);

  for (int l = 0; l < 4; ++l) {
    k_init_agg<<<cdiv_ll(NH, 256), 256, 0, stream>>>(HH, dinv, bb[l], AGG, N_NODES);
    k_edge_agg<<<cdiv_ll((long long)N_EDGES * 32, 256), 256, 0, stream>>>(
        srcI, dstI, dinv, HH, AGG, N_EDGES);
    k_fill<<<1, 256, 0, stream>>>(stats, 0.0f, 2 * HID);
    k_rational<<<cdiv_ll(N_NODES, RAT_ROWS), 128, 0, stream>>>(
        AGG, rnum, rden, assign, ACT, stats, N_NODES);
    k_bn<<<cdiv_ll(NH, 256), 256, 0, stream>>>(ACT, stats, gg[l], beta[l], Abf, N_NODES);
    if (l < 3) {
      k_wt<<<cdiv_ll(HID * HID, 256), 256, 0, stream>>>(W[l + 1], Wt);
      k_gemm_wmma<<<cdiv_ll(N_NODES, 128), 256, 0, stream>>>(Abf, Wt, HH, N_NODES);
    }
  }

  k_fill<<<cdiv_ll(NGROUP * HID, 256), 256, 0, stream>>>(pooled, 0.0f, NGROUP * HID);
  k_fill<<<1, 256, 0, stream>>>(counts, 0.0f, NGROUP);
  k_counts<<<cdiv_ll(N_NODES, 256), 256, 0, stream>>>(batch, counts, N_NODES);
  k_pool<<<cdiv_ll(NH, 256), 256, 0, stream>>>(batch, ACT, pooled, N_NODES);
  k_pdiv<<<cdiv_ll(NGROUP * HID, 256), 256, 0, stream>>>(pooled, counts);
  k_mlp_gelu<<<cdiv_ll(NGROUP * HID, 256), 256, 0, stream>>>(pooled, HW1, Hb1, z1);
  k_mlp_gelu<<<cdiv_ll(NGROUP * HID, 256), 256, 0, stream>>>(z1, HW2, Hb2, z2);
  k_mlp_out<<<cdiv_ll(NGROUP * OUT_DIM, 256), 256, 0, stream>>>(z2, HW3, Hb3, out);
}
